// model_30494267801652
// MI455X (gfx1250) — compile-verified, matrix operand order checked by reading, not run on hardware
//
#include <hip/hip_runtime.h>
#include <hip/hip_bf16.h>
#include <stdint.h>

#if __has_include(<hip/amd_detail/amd_gfx1250_TDM.h>)
#define TDM_SIX_ARG 1
#endif

typedef __attribute__((ext_vector_type(16))) _Float16 v16h;
typedef __attribute__((ext_vector_type(8)))  _Float16 v8h;
typedef __attribute__((ext_vector_type(4)))  _Float16 v4h;
typedef __attribute__((ext_vector_type(8)))  float    v8f;
typedef __attribute__((ext_vector_type(4)))  float    v4f;
typedef __attribute__((ext_vector_type(4)))  unsigned int u32x4;
typedef __attribute__((ext_vector_type(8)))  int          i32x8;
typedef __attribute__((ext_vector_type(4)))  int          i32x4;

#define NOBS   65536
#define DIM    64
#define HID    256
#define TM     64          // rows per workgroup
// padded LDS row strides (in halves); all rows 16B-aligned, banks spread
#define LIN    136
#define LW1    136
#define LW2    264
#define LW3    264
#define LH     264

// LDS partition offsets (in halves). Weight image = first 119,296 halves and is
// byte-identical to the prepacked image in d_ws (so TDM can do one flat copy).
#define OFF_W1T 0                       // 256 x 136
#define OFF_W2T 34816                   // 256 x 264
#define OFF_W3T 102400                  // 64  x 264
#define OFF_IN  119296                  // 64  x 136
#define OFF_H1  128000                  // 64  x 264
#define OFF_H2  144896                  // 64  x 264
#define SMEM_HALVES 161792              // 323,584 B dynamic LDS (<= 320KB/WGP)
#define WHALVES 119296
#define WBYTES  238592                  // f16 weight image bytes (= 29,824 x 8B)

__device__ __forceinline__ uint32_t pcg_hash(uint32_t x) {
    x = x * 747796405u + 2891336453u;
    uint32_t w = ((x >> ((x >> 28u) + 4u)) ^ x) * 277803737u;
    return (w >> 22u) ^ w;
}

__device__ __forceinline__ float gaussf(uint32_t seed) {
    uint32_t h1 = pcg_hash(seed);
    uint32_t h2 = pcg_hash(seed ^ 0x9E3779B9u);
    float u1 = (float)(h1 | 1u) * 2.3283064365386963e-10f;   // (0,1]
    float u2 = (float)h2 * 2.3283064365386963e-10f;
    float r = __builtin_sqrtf(-2.0f * __logf(u1));
    return r * __cosf(6.28318530718f * u2);
}

// 16x32 f16 fragment: two contiguous 16B chunks per lane (ds_load_b128 x2).
// Caller passes a per-wave/lane base pointer + compile-time-constant offset so
// the adds fold into the DS immediate offset field.
__device__ __forceinline__ v16h ld_frag(const _Float16* p) {
    v8h lo = *(const v8h*)(p);
    v8h hi = *(const v8h*)(p + 16);
    v16h f;
#pragma unroll
    for (int i = 0; i < 8; ++i) { f[i] = lo[i]; f[i + 8] = hi[i]; }
    return f;
}

__device__ __forceinline__ v8f wmma16(v16h a, v16h b, v8f c) {
    return __builtin_amdgcn_wmma_f32_16x16x32_f16(false, a, false, b,
                                                  (short)0, c, false, false);
}

// ---- prep: build f16 transposed + padded weight image in d_ws (runs once per launch)
__global__ void prep_weights(const float* __restrict__ W1,
                             const float* __restrict__ W2,
                             const float* __restrict__ W3,
                             _Float16* __restrict__ ws)
{
    int tid = blockIdx.x * blockDim.x + threadIdx.x;
    int nthr = gridDim.x * blockDim.x;
    for (int idx = tid; idx < 256 * LW1; idx += nthr) {          // W1t [n][k], k<128
        int n = idx / LW1, k = idx - n * LW1;
        ws[OFF_W1T + idx] = (k < 128) ? (_Float16)W1[k * 256 + n] : (_Float16)0.0f;
    }
    for (int idx = tid; idx < 256 * LW2; idx += nthr) {          // W2t [n][k], k<256
        int n = idx / LW2, k = idx - n * LW2;
        ws[OFF_W2T + idx] = (k < 256) ? (_Float16)W2[k * 256 + n] : (_Float16)0.0f;
    }
    for (int idx = tid; idx < 64 * LW3; idx += nthr) {           // W3t [n][k], k<256
        int n = idx / LW3, k = idx - n * LW3;
        ws[OFF_W3T + idx] = (k < 256) ? (_Float16)W3[k * 64 + n] : (_Float16)0.0f;
    }
}

template <bool USE_TDM>
__global__ void __launch_bounds__(256, 1)
interp_step(const float* __restrict__ X0g, const float* __restrict__ W1,
            const float* __restrict__ b1,  const float* __restrict__ W2,
            const float* __restrict__ b2,  const float* __restrict__ W3,
            const float* __restrict__ b3,  const _Float16* __restrict__ wsW,
            float* __restrict__ Xstate, int stepn, float s, float kc)
{
    extern __shared__ _Float16 smem[];
    _Float16* sW1t = smem + OFF_W1T;
    _Float16* sW2t = smem + OFF_W2T;
    _Float16* sW3t = smem + OFF_W3T;
    _Float16* sIn  = smem + OFF_IN;
    _Float16* sH1  = smem + OFF_H1;
    _Float16* sH2  = smem + OFF_H2;
    __shared__ float sB1e[HID];
    __shared__ float sB2[HID];
    __shared__ float sB3[DIM];

    const int tid     = threadIdx.x;
    const int lane    = tid & 31;
    const int wave    = tid >> 5;
    const int rowBase = blockIdx.x * TM;
    const int sample  = blockIdx.y;

    float* Xs = Xstate + (size_t)sample * NOBS * DIM;
    const float* Xin = (stepn == 0) ? X0g : Xs;

    // ---- weight staging ----
    if constexpr (USE_TDM) {
        // One Tensor-Data-Mover DMA per WG: copy the whole 238,592B f16 weight
        // image (1-D tile of 29,824 x 8B units) from d_ws into LDS. Wave 0 only.
        if (tid < 32) {
            uint32_t ldsOff = (uint32_t)(uintptr_t)(void*)smem;   // flat LDS addr低32 = LDS offset
            uint64_t ga = (uint64_t)(uintptr_t)(const void*)wsW;
            const uint32_t e8 = WBYTES >> 3;                      // 29,824 8B elements
            u32x4 g0 = {};
            g0[0] = 1u;                                           // count=1, user D#
            g0[1] = ldsOff;                                       // lds_addr
            g0[2] = (uint32_t)ga;                                 // global_addr[31:0]
            g0[3] = ((uint32_t)(ga >> 32) & 0x01FFFFFFu) | (2u << 30); // addr[56:32], type=2
            i32x8 g1 = {};
            g1[0] = (int)(3u << 16);                              // data_size=3 (8B)
            g1[1] = (int)((e8 & 0xFFFFu) << 16);                  // tensor_dim0[15:0]
            g1[2] = (int)((e8 >> 16) | (1u << 16));               // tensor_dim0[31:16], tensor_dim1=1
            g1[3] = (int)((e8 & 0xFFFFu) << 16);                  // tile_dim0 = e8
            g1[4] = 1;                                            // tile_dim1=1, tile_dim2=0
            g1[5] = (int)e8;                                      // tensor_dim0_stride[31:0]
            i32x4 z4 = {};
#ifdef TDM_SIX_ARG
            i32x8 z8 = {};
            __builtin_amdgcn_tensor_load_to_lds(g0, g1, z4, z4, z8, 0);
#else
            __builtin_amdgcn_tensor_load_to_lds(g0, g1, z4, z4, 0);
#endif
            __builtin_amdgcn_s_wait_tensorcnt(0);
        }
    } else {
        // Fallback: convert fp32 weights -> f16 transposed in LDS.
        for (int idx = tid; idx < 128 * HID; idx += 256) {
            int k = idx >> 8, n = idx & 255;
            sW1t[n * LW1 + k] = (_Float16)W1[idx];
        }
        for (int idx = tid; idx < HID * HID; idx += 256) {
            int k = idx >> 8, n = idx & 255;
            sW2t[n * LW2 + k] = (_Float16)W2[idx];
        }
        for (int idx = tid; idx < HID * DIM; idx += 256) {
            int k = idx >> 6, n = idx & 63;
            sW3t[n * LW3 + k] = (_Float16)W3[idx];
        }
    }

    // ---- biases (fold s * W1[128,:] into b1) ----
    if (tid < HID) {
        sB1e[tid] = b1[tid] + s * W1[128 * HID + tid];
        sB2[tid]  = b2[tid];
    }
    if (tid < DIM) sB3[tid] = b3[tid];

    // ---- input tile [x(64) | x0(64)] as f16, float4 -> half4 ----
    for (int idx = tid; idx < TM * DIM / 4; idx += 256) {
        int r = idx >> 4, c4 = (idx & 15) << 2;
        int row = rowBase + r;
        v4f xv  = *(const v4f*)(Xin + row * DIM + c4);
        v4f x0v = *(const v4f*)(X0g + row * DIM + c4);
        v4h xh, x0h;
#pragma unroll
        for (int i = 0; i < 4; ++i) { xh[i] = (_Float16)xv[i]; x0h[i] = (_Float16)x0v[i]; }
        *(v4h*)(sIn + r * LIN + c4)       = xh;
        *(v4h*)(sIn + r * LIN + DIM + c4) = x0h;
    }
    __syncthreads();

    // wave tiling: 2 m-tiles x 4 n-tiles (GEMM1/2), 2 m-tiles x 1 n-tile (GEMM3)
    const int mtBase  = (wave & 1) * 32;
    const int ntCol   = (wave >> 1) * 64;    // GEMM1/2 column base
    const int ntCol3  = (wave >> 1) * 16;    // GEMM3 column base
    const int colLane = lane & 15;
    const int g8      = (lane >> 4) << 3;    // upper half-wave: +8 halves in K
    const int rowSel8 = (lane >> 4) << 3;    // upper half-wave: +8 rows in C/D

    // ================= GEMM1: [64,128] x [128,256] -> relu -> sH1 =================
    {
        const _Float16* aB = sIn  + (mtBase + colLane) * LIN + g8;
        const _Float16* bB = sW1t + (ntCol  + colLane) * LW1 + g8;
        v8f acc[2][4] = {};
#pragma unroll
        for (int kt = 0; kt < 128; kt += 32) {
            v16h a0 = ld_frag(aB + kt);
            v16h a1 = ld_frag(aB + 16 * LIN + kt);
#pragma unroll
            for (int nj = 0; nj < 4; ++nj) {
                v16h bf = ld_frag(bB + nj * 16 * LW1 + kt);
                acc[0][nj] = wmma16(a0, bf, acc[0][nj]);
                acc[1][nj] = wmma16(a1, bf, acc[1][nj]);
            }
        }
        _Float16* st = sH1 + (mtBase + rowSel8) * LH + ntCol + colLane;
#pragma unroll
        for (int nj = 0; nj < 4; ++nj) {
            float bv = sB1e[ntCol + nj * 16 + colLane];
#pragma unroll
            for (int mi = 0; mi < 2; ++mi)
#pragma unroll
                for (int r = 0; r < 8; ++r) {
                    float v = acc[mi][nj][r] + bv;
                    v = v > 0.0f ? v : 0.0f;
                    st[(mi * 16 + r) * LH + nj * 16] = (_Float16)v;
                }
        }
    }
    __syncthreads();

    // ================= GEMM2: [64,256] x [256,256] -> relu -> sH2 =================
    {
        const _Float16* aB = sH1  + (mtBase + colLane) * LH  + g8;
        const _Float16* bB = sW2t + (ntCol  + colLane) * LW2 + g8;
        v8f acc[2][4] = {};
#pragma unroll
        for (int kt = 0; kt < 256; kt += 32) {
            v16h a0 = ld_frag(aB + kt);
            v16h a1 = ld_frag(aB + 16 * LH + kt);
#pragma unroll
            for (int nj = 0; nj < 4; ++nj) {
                v16h bf = ld_frag(bB + nj * 16 * LW2 + kt);
                acc[0][nj] = wmma16(a0, bf, acc[0][nj]);
                acc[1][nj] = wmma16(a1, bf, acc[1][nj]);
            }
        }
        _Float16* st = sH2 + (mtBase + rowSel8) * LH + ntCol + colLane;
#pragma unroll
        for (int nj = 0; nj < 4; ++nj) {
            float bv = sB2[ntCol + nj * 16 + colLane];
#pragma unroll
            for (int mi = 0; mi < 2; ++mi)
#pragma unroll
                for (int r = 0; r < 8; ++r) {
                    float v = acc[mi][nj][r] + bv;
                    v = v > 0.0f ? v : 0.0f;
                    st[(mi * 16 + r) * LH + nj * 16] = (_Float16)v;
                }
        }
    }
    __syncthreads();

    // ============ GEMM3: [64,256] x [256,64] -> drift; fused SDE update ============
    {
        const _Float16* aB = sH2  + (mtBase + colLane) * LH  + g8;
        const _Float16* bB = sW3t + (ntCol3 + colLane) * LW3 + g8;
        v8f acc[2] = {};
#pragma unroll
        for (int kt = 0; kt < 256; kt += 32) {
            v16h bf = ld_frag(bB + kt);
            v16h a0 = ld_frag(aB + kt);
            v16h a1 = ld_frag(aB + 16 * LH + kt);
            acc[0] = wmma16(a0, bf, acc[0]);
            acc[1] = wmma16(a1, bf, acc[1]);
        }
        // X += ds*drift + kc*(s*drift - X + X0) + sqrt(ds)*eta  (bias folded in)
        const float bv = sB3[ntCol3 + colLane];
        const uint32_t mix = ((uint32_t)stepn * 2u + (uint32_t)sample) * 0x9E3779B9u;
#pragma unroll
        for (int mi = 0; mi < 2; ++mi)
#pragma unroll
            for (int r = 0; r < 8; ++r) {
                int row = rowBase + mtBase + mi * 16 + r + rowSel8;
                int off = row * DIM + ntCol3 + colLane;
                float drift = acc[mi][r] + bv;
                float x  = Xin[off];
                float x0 = X0g[off];
                float eta = gaussf(mix ^ (uint32_t)off);
                float xn = x + 0.015625f * drift
                             + kc * (s * drift - x + x0)
                             + 0.125f * eta;
                Xs[off] = xn;
            }
    }
}

extern "C" void kernel_launch(void* const* d_in, const int* in_sizes, int n_in,
                              void* d_out, int out_size, void* d_ws, size_t ws_size,
                              hipStream_t stream)
{
    (void)in_sizes; (void)n_in; (void)out_size;
    const float* X0 = (const float*)d_in[0];
    const float* W1 = (const float*)d_in[1];
    const float* b1 = (const float*)d_in[2];
    const float* W2 = (const float*)d_in[3];
    const float* b2 = (const float*)d_in[4];
    const float* W3 = (const float*)d_in[5];
    const float* b3 = (const float*)d_in[6];
    float* Xstate = (float*)d_out;            // [2, 65536, 64] — doubles as SDE state
    _Float16* wsW = (_Float16*)d_ws;

    const bool useTdm = (ws_size >= (size_t)WBYTES) && (d_ws != nullptr);

    dim3 grid(NOBS / TM, 2);
    dim3 block(256);
    size_t smem = (size_t)SMEM_HALVES * sizeof(_Float16);
    const float ds = 1.0f / 64.0f;

    if (useTdm)
        prep_weights<<<dim3(512), dim3(256), 0, stream>>>(W1, W2, W3, wsW);

    for (int n = 0; n < 64; ++n) {
        float s = (float)n / 64.0f;
        float kc = 0.0f;
        if (n > 0) {
            float sg = 1.0f - s;
            kc = ds * 0.5f * (1.0f - sg * sg) / (s * sg);
        }
        if (useTdm)
            interp_step<true><<<grid, block, smem, stream>>>(
                X0, W1, b1, W2, b2, W3, b3, wsW, Xstate, n, s, kc);
        else
            interp_step<false><<<grid, block, smem, stream>>>(
                X0, W1, b1, W2, b2, W3, b3, wsW, Xstate, n, s, kc);
    }
}
